// VectorQuantizedVAE_2156073583343
// MI455X (gfx1250) — compile-verified, matrix operand order checked by reading, not run on hardware
//
#include <hip/hip_runtime.h>
#include <hip/hip_bf16.h>
#include <cstdint>

// ---------------------------------------------------------------------------
// VQ-VAE forward for MI455X (gfx1250).
//  - conv / convt / VQ distances via v_wmma_f32_16x16x32_bf16
//  - weight panels DMA'd to LDS by the Tensor Data Mover (TENSORcnt),
//    double-buffered across taps; activations gathered+converted to bf16 in
//    LDS, double-buffered per K-step (ASYNCcnt for raw copies as fallback)
//  - bias/ReLU/tanh fused into GEMM epilogues; BN = 2 bandwidth passes
// ---------------------------------------------------------------------------

typedef __attribute__((ext_vector_type(16))) __bf16 v16bf;
typedef __attribute__((ext_vector_type(8)))  __bf16 v8bf;
typedef __attribute__((ext_vector_type(8)))  float  v8f;
typedef __attribute__((ext_vector_type(4)))  unsigned v4u;
typedef __attribute__((ext_vector_type(4)))  int  v4i;
typedef __attribute__((ext_vector_type(8)))  int  v8i;

#define EPI_BIAS 0
#define EPI_RELU 1
#define EPI_TANH 2

#define USE_ASYNC_LDS 1

#if defined(__gfx1250__) && __has_builtin(__builtin_amdgcn_tensor_load_to_lds) && \
    __has_builtin(__builtin_amdgcn_s_wait_tensorcnt)
  #define HAVE_TDM 1
  #if __clang_major__ >= 23
    #define TDM_ARITY6 1
  #endif
#endif

#if USE_ASYNC_LDS
__device__ __forceinline__ void async_b128(unsigned lds_off, const void* gaddr) {
  asm volatile("global_load_async_to_lds_b128 %0, %1, off"
               :: "v"(lds_off), "v"(gaddr) : "memory");
}
__device__ __forceinline__ void async_wait() {
  asm volatile("s_wait_asynccnt 0" ::: "memory");
}
#endif

__device__ __forceinline__ void tensor_wait() {
#if defined(HAVE_TDM)
  __builtin_amdgcn_s_wait_tensorcnt(0);
#endif
}

#if defined(HAVE_TDM)
// TDM 2D tile load: tileRows x tileCols bf16 elements, row stride strideElems,
// packed contiguously into LDS at ldsAddr. tensorRows < tileRows => OOB rows
// read as zero (D# OOB rule).  D# bit layout per CDNA5 ISA ch.8.
__device__ __forceinline__ void tdm_load_2d(unsigned ldsAddr, const void* gsrc,
                                            int tileCols, int tileRows,
                                            int strideElems, int tensorRows) {
  unsigned long long ga = (unsigned long long)(uintptr_t)gsrc;
  v4u g0;
  g0[0] = 1u;                                        // count=1, user D#
  g0[1] = ldsAddr;                                   // lds_addr
  g0[2] = (unsigned)(ga & 0xFFFFFFFFu);              // global_addr[31:0]
  g0[3] = (unsigned)((ga >> 32) & 0x01FFFFFFu) | 0x80000000u; // [56:32] | type=2
  v8i g1;
  g1[0] = (int)(1u << 16);                           // data_size=1 (2 bytes)
  g1[1] = (int)(((unsigned)tileCols & 0xFFFFu) << 16);          // tensor_dim0 lo
  g1[2] = (int)((((unsigned)tileCols >> 16) & 0xFFFFu) |
                (((unsigned)tensorRows & 0xFFFFu) << 16));      // dim0 hi | dim1 lo
  g1[3] = (int)((((unsigned)tensorRows >> 16) & 0xFFFFu) |
                (((unsigned)tileCols & 0xFFFFu) << 16));        // dim1 hi | tile_dim0
  g1[4] = (int)((unsigned)tileRows & 0xFFFFu);                  // tile_dim1, tile_dim2=0
  g1[5] = (int)strideElems;                                     // tensor_dim0_stride lo
  g1[6] = 0;                                                    // stride0 hi | stride1 lo
  g1[7] = 0;                                                    // stride1 hi
  v4i gz = {0, 0, 0, 0};
#if defined(TDM_ARITY6)
  v8i gz8 = {0, 0, 0, 0, 0, 0, 0, 0};
  __builtin_amdgcn_tensor_load_to_lds(g0, g1, gz, gz, gz8, 0);
#else
  __builtin_amdgcn_tensor_load_to_lds(g0, g1, gz, gz, 0);
#endif
}
#endif

// 16x32 bf16 WMMA fragment from an LDS row (K-contiguous). ISA 7.12.2:
// lanes 0-15 hold K {0..7,16..23}, lanes 16-31 hold K {8..15,24..31}.
__device__ __forceinline__ v16bf lds_frag(const __bf16* rowBase, int half) {
  v8bf lo = *(const v8bf*)(rowBase + half * 8);
  v8bf hi = *(const v8bf*)(rowBase + 16 + half * 8);
  v16bf r;
#pragma unroll
  for (int i = 0; i < 8; ++i) { r[i] = lo[i]; r[i + 8] = hi[i]; }
  return r;
}

__device__ __forceinline__ v8f bf16_wmma(v16bf a, v16bf b, v8f c) {
  return __builtin_amdgcn_wmma_f32_16x16x32_bf16(false, a, false, b, (short)0, c, false, false);
}

// ---------------------------------------------------------------------------
// Implicit-GEMM conv / conv-transpose (lhs-dilation `ldil`, pre-flipped w).
// Block = 256 thr = 8 waves; 128(M) x 64(Cout) tile; wave (wr,wc) owns 32x32
// (2 A frags x 2 B frags -> 4 WMMA per K-step). Weight panel per tap loaded
// by TDM into LDS (double-buffered across taps); A tile double-buffered per
// K-step. OW / OH*OW are powers of two -> shift decode, hoisted out of loop.
// ---------------------------------------------------------------------------
__global__ __launch_bounds__(256) void k_conv_wmma(
    const float* __restrict__ in, const __bf16* __restrict__ W2,
    const float* __restrict__ bias, float* __restrict__ out,
    int N, int Cin, int IH, int IW, int Cout, int OH, int OW,
    int owLog, int ohwLog,
    int k, int stride, int pad, int ldil, int KcPad, int epi)
{
  __shared__ __align__(16) __bf16 As[2][128 * 32];
  __shared__ __align__(16) __bf16 Bp[2][64 * 256];

  const int tid  = threadIdx.x;
  const int wave = tid >> 5, lane = tid & 31;
  const int half = lane >> 4, lm = lane & 15;
  const int M  = N * OH * OW;
  const int m0 = blockIdx.x * 128, n0 = blockIdx.y * 64;
  const int taps = k * k, ksteps = KcPad >> 5;
  const int IHW = IH * IW;

  // loop-invariant decode of the two A-staging rows owned by this thread
  int aRow[2], aSeg[2], aNb[2], aOy[2], aOx[2]; bool aOk[2];
#pragma unroll
  for (int cc = 0; cc < 2; ++cc) {
    int ch = tid + 256 * cc;
    aRow[cc] = ch >> 2; aSeg[cc] = (ch & 3) * 8;
    int gm = m0 + aRow[cc];
    aOk[cc] = gm < M;
    int gmc = aOk[cc] ? gm : 0;
    aNb[cc] = gmc >> ohwLog;
    int rem = gmc & ((1 << ohwLog) - 1);
    aOy[cc] = rem >> owLog;
    aOx[cc] = rem & ((1 << owLog) - 1);
  }

  auto loadA = [&](int tap, int ks, int buf) {
    int ky = tap / k, kx = tap - (tap / k) * k;
    int kc0 = ks << 5;
#pragma unroll
    for (int cc = 0; cc < 2; ++cc) {
      int ty = aOy[cc] * stride - pad + ky;
      int tx = aOx[cc] * stride - pad + kx;
      int iy = ty / ldil, ix = tx / ldil;
      bool pix = aOk[cc] && (ty >= 0) && (tx >= 0) && (ty % ldil == 0) &&
                 (tx % ldil == 0) && (iy < IH) && (ix < IW);
      const float* src = in + ((long)aNb[cc] * Cin + kc0 + aSeg[cc]) * IHW +
                         (long)iy * IW + ix;
      __bf16* dst = &As[buf][aRow[cc] * 32 + aSeg[cc]];
#pragma unroll
      for (int j = 0; j < 8; ++j) {
        int c = kc0 + aSeg[cc] + j;
        float v = (pix && c < Cin) ? src[(long)j * IHW] : 0.f;
        dst[j] = (__bf16)v;
      }
    }
  };

  auto loadPanel = [&](int tap, int pb) {
    const __bf16* base = W2 + ((long)tap * Cout + n0) * KcPad;
#if defined(HAVE_TDM)
    if (wave == 0) {
      int rowsLeft = Cout - n0;
      tdm_load_2d((unsigned)(uintptr_t)&Bp[pb][0], base, KcPad, 64, KcPad, rowsLeft);
    }
#else
    int segsPerRow = KcPad >> 3;
    int rowShift = __builtin_ctz(segsPerRow);
    int chunks = 64 * segsPerRow;
    for (int ch = tid; ch < chunks; ch += 256) {
      int row = ch >> rowShift;
      int seg = (ch & (segsPerRow - 1)) * 8;
      __bf16* dst = &Bp[pb][row * KcPad + seg];
      if (n0 + row < Cout) {
#if USE_ASYNC_LDS
        async_b128((unsigned)(uintptr_t)dst, base + (long)row * KcPad + seg);
#else
        *(v8bf*)dst = *(const v8bf*)(base + (long)row * KcPad + seg);
#endif
      } else {
#pragma unroll
        for (int j = 0; j < 8; ++j) dst[j] = (__bf16)0.f;
      }
    }
#endif
  };

  const int wr = wave & 3, wc = wave >> 2;
  v8f z8 = {};
  v8f acc00 = z8, acc01 = z8, acc10 = z8, acc11 = z8;

  loadPanel(0, 0);
  loadA(0, 0, 0);
#if USE_ASYNC_LDS
  async_wait();
#endif
  tensor_wait();
  __syncthreads();

  int abuf = 0, pbuf = 0;
  for (int tap = 0; tap < taps; ++tap) {
    for (int ks = 0; ks < ksteps; ++ks) {
      bool lastKs  = (ks == ksteps - 1);
      bool lastAll = lastKs && (tap == taps - 1);
      if (ks == 0 && tap + 1 < taps) loadPanel(tap + 1, pbuf ^ 1);  // DMA next tap
      if (!lastAll) loadA(lastKs ? tap + 1 : tap, lastKs ? 0 : ks + 1, abuf ^ 1);

      int kc0 = ks << 5;
      v16bf a0 = lds_frag(&As[abuf][(wr * 32      + lm) * 32], half);
      v16bf a1 = lds_frag(&As[abuf][(wr * 32 + 16 + lm) * 32], half);
      v16bf b0 = lds_frag(&Bp[pbuf][(wc * 32      + lm) * KcPad + kc0], half);
      v16bf b1 = lds_frag(&Bp[pbuf][(wc * 32 + 16 + lm) * KcPad + kc0], half);
      acc00 = bf16_wmma(a0, b0, acc00);
      acc01 = bf16_wmma(a0, b1, acc01);
      acc10 = bf16_wmma(a1, b0, acc10);
      acc11 = bf16_wmma(a1, b1, acc11);

#if USE_ASYNC_LDS
      async_wait();
#endif
      if (lastKs) tensor_wait();
      __syncthreads();
      abuf ^= 1;
      if (lastKs) pbuf ^= 1;
    }
  }

  // Epilogue. C/D layout: lanes 0-15 -> M=r, lanes 16-31 -> M=r+8; N = lane&15.
#pragma unroll
  for (int at = 0; at < 2; ++at) {
#pragma unroll
    for (int bt = 0; bt < 2; ++bt) {
      v8f acc = at ? (bt ? acc11 : acc10) : (bt ? acc01 : acc00);
      int co = n0 + wc * 32 + bt * 16 + lm;
      if (co >= Cout) continue;
      float bv = bias[co];
#pragma unroll
      for (int r = 0; r < 8; ++r) {
        int gm = m0 + wr * 32 + at * 16 + r + half * 8;
        if (gm >= M) continue;
        int nb = gm >> ohwLog;
        int rem = gm & ((1 << ohwLog) - 1);
        int oy = rem >> owLog, ox = rem & ((1 << owLog) - 1);
        float v = acc[r] + bv;
        if (epi == EPI_RELU) v = fmaxf(v, 0.f);
        else if (epi == EPI_TANH) v = tanhf(v);
        out[(((long)nb * Cout + co) * OH + oy) * OW + ox] = v;
      }
    }
  }
}

// ---------------------------------------------------------------------------
// VQ: scores = |W|^2 - 2 z.W via WMMA; argmin over 512 codes.
// Block: 16 pixels x 512 codes; wave w owns codes w*64..w*64+63 (4 tiles).
// Codebook K-slice (512x32, row stride 256) DMA'd by TDM.
// ---------------------------------------------------------------------------
__global__ __launch_bounds__(256) void k_vq_argmin(
    const float* __restrict__ z, const __bf16* __restrict__ cbT,
    const float* __restrict__ wsq, int* __restrict__ idxOut,
    int NP, int C, int HW, int CS)
{
  (void)CS;
  __shared__ __align__(16) __bf16 As[16 * 32];
  __shared__ __align__(16) __bf16 Bs[512 * 32];
  __shared__ float redS[16][8];
  __shared__ int   redI[16][8];

  const int tid  = threadIdx.x;
  const int wave = tid >> 5, lane = tid & 31;
  const int half = lane >> 4, lm = lane & 15;
  const int p0 = blockIdx.x * 16;
  const int ksteps = C >> 5;

  v8f z8 = {};
  v8f acc[4] = { z8, z8, z8, z8 };

  for (int ks = 0; ks < ksteps; ++ks) {
    int kc0 = ks << 5;
    if (ks) __syncthreads();           // previous compute done before overwrite
    { // A: 16 pixels x 32 channels (channel-strided gather from NCHW z)
      int e = tid * 2;
      int row = e >> 5, col = e & 31;
      int gp = p0 + row;
      int nb = gp / HW, yx = gp - nb * HW;
      const float* src = &z[((long)nb * C) * HW + yx];
      __bf16* dst = &As[row * 32 + col];
      dst[0] = (__bf16)src[(long)(kc0 + col    ) * HW];
      dst[1] = (__bf16)src[(long)(kc0 + col + 1) * HW];
    }
#if defined(HAVE_TDM)
    if (wave == 0)
      tdm_load_2d((unsigned)(uintptr_t)&Bs[0], cbT + kc0, 32, 512, C, 512);
#else
#pragma unroll
    for (int q = 0; q < 8; ++q) {
      int ch = tid + 256 * q;
      int row = ch >> 2, seg = (ch & 3) * 8;
      __bf16* dst = &Bs[row * 32 + seg];
#if USE_ASYNC_LDS
      async_b128((unsigned)(uintptr_t)dst, cbT + (long)row * C + kc0 + seg);
#else
      *(v8bf*)dst = *(const v8bf*)(cbT + (long)row * C + kc0 + seg);
#endif
    }
#endif
#if USE_ASYNC_LDS
    async_wait();
#endif
    tensor_wait();
    __syncthreads();

    v16bf a = lds_frag(&As[lm * 32], half);
#pragma unroll
    for (int t = 0; t < 4; ++t) {
      v16bf b = lds_frag(&Bs[(wave * 64 + t * 16 + lm) * 32], half);
      acc[t] = bf16_wmma(a, b, acc[t]);
    }
  }
  __syncthreads();

  float bs[8]; int bi[8];
#pragma unroll
  for (int r = 0; r < 8; ++r) { bs[r] = 3.0e38f; bi[r] = 0; }
#pragma unroll
  for (int t = 0; t < 4; ++t) {
    int code = wave * 64 + t * 16 + lm;
    float w2 = wsq[code];
#pragma unroll
    for (int r = 0; r < 8; ++r) {
      float s = w2 - 2.0f * acc[t][r];
      if (s < bs[r]) { bs[r] = s; bi[r] = code; }
    }
  }
#pragma unroll
  for (int off = 1; off < 16; off <<= 1) {
#pragma unroll
    for (int r = 0; r < 8; ++r) {
      float os = __shfl_xor(bs[r], off, 32);
      int   oi = __shfl_xor(bi[r], off, 32);
      if (os < bs[r]) { bs[r] = os; bi[r] = oi; }
    }
  }
  if (lm == 0) {
#pragma unroll
    for (int r = 0; r < 8; ++r) {
      int row = r + half * 8;
      redS[row][wave] = bs[r];
      redI[row][wave] = bi[r];
    }
  }
  __syncthreads();
  if (tid < 16) {
    float m = 3.0e38f; int mi = 0;
    for (int w = 0; w < 8; ++w)
      if (redS[tid][w] < m) { m = redS[tid][w]; mi = redI[tid][w]; }
    if (p0 + tid < NP) idxOut[p0 + tid] = mi;
  }
}

// ------------------------------ utility kernels ----------------------------

__global__ __launch_bounds__(256) void k_prep_w(
    const float* __restrict__ w, __bf16* __restrict__ w2,
    int Cout, int Cin, int k, int KcPad, int transposed)
{
  int n = k * k * Cout * KcPad;
  for (int id = blockIdx.x * 256 + threadIdx.x; id < n; id += gridDim.x * 256) {
    int tap = id / (Cout * KcPad);
    int rem = id - tap * (Cout * KcPad);
    int o = rem / KcPad, i = rem - o * KcPad;
    int ky = tap / k, kx = tap - ky * k;
    float v = 0.f;
    if (i < Cin) {
      if (transposed) v = w[(((long)i * Cout + o) * k + (k - 1 - ky)) * k + (k - 1 - kx)];
      else            v = w[(((long)o * Cin + i) * k + ky) * k + kx];
    }
    w2[id] = (__bf16)v;
  }
}

__global__ __launch_bounds__(256) void k_prep_cb(
    const float* __restrict__ cb, __bf16* __restrict__ cbT,
    float* __restrict__ wsq, int CS, int CD)
{
  int code = blockIdx.x, tid = threadIdx.x;
  __shared__ float red[256];
  float s = 0.f;
  for (int c = tid; c < CD; c += 256) {
    float v = cb[(long)code * CD + c];
    cbT[(long)code * CD + c] = (__bf16)v;
    s += v * v;
  }
  red[tid] = s; __syncthreads();
  for (int st = 128; st > 0; st >>= 1) { if (tid < st) red[tid] += red[tid + st]; __syncthreads(); }
  if (tid == 0) wsq[code] = red[0];
}

__global__ __launch_bounds__(256) void k_bn_stats(
    const float* __restrict__ x, float* __restrict__ stats,
    int N, int C, int HW, int prerelu)
{
  int c = blockIdx.x, tid = threadIdx.x;
  __shared__ float rs[256], rq[256];
  float s = 0.f, q = 0.f;
  long tot = (long)N * HW;
  for (long i = tid; i < tot; i += 256) {
    long n = i / HW, j = i - n * HW;
    float v = x[((n * C + c) * (long)HW) + j];
    if (prerelu) v = fmaxf(v, 0.f);
    s += v; q += v * v;
  }
  rs[tid] = s; rq[tid] = q; __syncthreads();
  for (int st = 128; st > 0; st >>= 1) {
    if (tid < st) { rs[tid] += rs[tid + st]; rq[tid] += rq[tid + st]; }
    __syncthreads();
  }
  if (tid == 0) {
    float m = rs[0] / (float)tot;
    float var = rq[0] / (float)tot - m * m;
    stats[c * 2] = m;
    stats[c * 2 + 1] = rsqrtf(var + 1e-5f);
  }
}

__global__ __launch_bounds__(256) void k_bn_apply(
    const float* __restrict__ x, float* __restrict__ y,
    const float* __restrict__ stats, const float* __restrict__ g,
    const float* __restrict__ b, long total, int C, int HW,
    int prerelu, int postrelu)
{
  for (long i = blockIdx.x * 256L + threadIdx.x; i < total; i += (long)gridDim.x * 256) {
    int c = (int)((i / HW) % C);
    float v = x[i];
    if (prerelu) v = fmaxf(v, 0.f);
    v = (v - stats[c * 2]) * stats[c * 2 + 1] * g[c] + b[c];
    if (postrelu) v = fmaxf(v, 0.f);
    y[i] = v;
  }
}

__global__ __launch_bounds__(256) void k_add(
    const float* __restrict__ a, const float* __restrict__ b,
    float* __restrict__ y, long n)
{
  for (long i = blockIdx.x * 256L + threadIdx.x; i < n; i += (long)gridDim.x * 256)
    y[i] = a[i] + b[i];
}

__global__ __launch_bounds__(256) void k_zero(float* p, int n)
{
  int i = blockIdx.x * 256 + threadIdx.x;
  if (i < n) p[i] = 0.f;
}

__global__ __launch_bounds__(256) void k_vq_gather(
    const float* __restrict__ z, const float* __restrict__ cb,
    const int* __restrict__ idx, float* __restrict__ q,
    float* __restrict__ acc, long total, int C, int HW)
{
  __shared__ float red[256];
  float s = 0.f;
  long chw = (long)C * HW;
  for (long i = blockIdx.x * 256L + threadIdx.x; i < total; i += (long)gridDim.x * 256) {
    long nb = i / chw; long rem = i - nb * chw;
    int c = (int)(rem / HW); int yx = (int)(rem - (long)c * HW);
    int p = (int)(nb * HW + yx);
    float qv = cb[(long)idx[p] * C + c];
    float d = z[i] - qv;
    q[i] = qv;
    s += d * d;
  }
  red[threadIdx.x] = s; __syncthreads();
  for (int st = 128; st > 0; st >>= 1) {
    if (threadIdx.x < st) red[threadIdx.x] += red[threadIdx.x + st];
    __syncthreads();
  }
  if (threadIdx.x == 0) atomicAdd(acc, red[0]);
}

__global__ void k_final(const float* acc, float* out, float scale)
{
  if (threadIdx.x == 0 && blockIdx.x == 0) out[0] = acc[0] * scale;
}

// --------------------------------- host ------------------------------------

enum {
  IN_X = 0,
  E_CONV1_W, E_CONV1_B, E_BN1_G, E_BN1_B,
  E_CONV2_W, E_CONV2_B, E_BN2_G, E_BN2_B,
  E_CONV3_W, E_CONV3_B, E_BN3_G, E_BN3_B,
  E_CONV4_W, E_CONV4_B,
  CODEBOOK,
  D_BN0_G, D_BN0_B,
  D_CONVT1_W, D_CONVT1_B, D_BN1_G, D_BN1_B,
  D_CONVT2_W, D_CONVT2_B, D_BN2_G, D_BN2_B,
  D_CONVT3_W, D_CONVT3_B, D_BN3_G, D_BN3_B,
  D_CONVT4_W, D_CONVT4_B,
  ER1_BN1_G, ER1_BN1_B, ER1_C1_W, ER1_C1_B, ER1_BN2_G, ER1_BN2_B, ER1_C2_W, ER1_C2_B,
  ER2_BN1_G, ER2_BN1_B, ER2_C1_W, ER2_C1_B, ER2_BN2_G, ER2_BN2_B, ER2_C2_W, ER2_C2_B,
  DR1_BN1_G, DR1_BN1_B, DR1_C1_W, DR1_C1_B, DR1_BN2_G, DR1_BN2_B, DR1_C2_W, DR1_C2_B,
  DR2_BN1_G, DR2_BN1_B, DR2_C1_W, DR2_C1_B, DR2_BN2_G, DR2_BN2_B, DR2_C2_W, DR2_C2_B,
};

extern "C" void kernel_launch(void* const* d_in, const int* in_sizes, int n_in,
                              void* d_out, int out_size, void* d_ws, size_t ws_size,
                              hipStream_t stream)
{
  (void)in_sizes; (void)n_in; (void)out_size; (void)ws_size;

  constexpr size_t SZ_BIG = 134217728;                 // 32*256*64*64 f32
  constexpr size_t SZ_Z   = 33554432;                  // 32*256*32*32 f32
  constexpr size_t OFF_A     = 0;
  constexpr size_t OFF_B     = OFF_A + SZ_BIG;
  constexpr size_t OFF_C     = OFF_B + SZ_BIG;
  constexpr size_t OFF_Q     = OFF_C + SZ_Z;
  constexpr size_t OFF_IDX   = OFF_Q + SZ_Z;
  constexpr size_t OFF_STATS = OFF_IDX + 131072;
  constexpr size_t OFF_WSQ   = OFF_STATS + 4096;
  constexpr size_t OFF_LOSS  = OFF_WSQ + 2048;
  constexpr size_t OFF_WB    = OFF_LOSS + 256;

  char* ws = (char*)d_ws;
  auto P = [&](int i) { return (const float*)d_in[i]; };
  float* bufA  = (float*)(ws + OFF_A);
  float* bufB  = (float*)(ws + OFF_B);
  float* bufC  = (float*)(ws + OFF_C);
  float* bufQ  = (float*)(ws + OFF_Q);
  int*   idxb  = (int*)(ws + OFF_IDX);
  float* stats = (float*)(ws + OFF_STATS);
  float* wsq   = (float*)(ws + OFF_WSQ);
  float* loss  = (float*)(ws + OFF_LOSS);
  float* outp  = (float*)d_out;

  size_t wo = OFF_WB;
  auto walloc = [&](int k, int Cout, int KcPad) {
    size_t o = wo;
    wo += (size_t)k * k * Cout * KcPad * sizeof(__bf16);
    wo = (wo + 255) & ~(size_t)255;
    return o;
  };
  size_t w_e1 = walloc(4, 64, 32),  w_e2 = walloc(3, 128, 64);
  size_t w_e3 = walloc(3, 256, 128), w_e4 = walloc(4, 256, 256);
  size_t w_er1c1 = walloc(3, 256, 256), w_er1c2 = walloc(1, 256, 256);
  size_t w_er2c1 = walloc(3, 256, 256), w_er2c2 = walloc(1, 256, 256);
  size_t w_dr1c1 = walloc(3, 256, 256), w_dr1c2 = walloc(1, 256, 256);
  size_t w_dr2c1 = walloc(3, 256, 256), w_dr2c2 = walloc(1, 256, 256);
  size_t w_d1 = walloc(4, 256, 256), w_d2 = walloc(3, 128, 256);
  size_t w_d3 = walloc(3, 64, 128),  w_d4 = walloc(4, 3, 64);
  size_t w_cb = wo;

  auto wprep = [&](int pi, size_t off, int Cout, int Cin, int k, int KcPad, int tr) {
    int n = k * k * Cout * KcPad;
    k_prep_w<<<dim3((n + 255) / 256), 256, 0, stream>>>(
        P(pi), (__bf16*)(ws + off), Cout, Cin, k, KcPad, tr);
  };
  wprep(E_CONV1_W, w_e1, 64, 3, 4, 32, 0);
  wprep(E_CONV2_W, w_e2, 128, 64, 3, 64, 0);
  wprep(E_CONV3_W, w_e3, 256, 128, 3, 128, 0);
  wprep(E_CONV4_W, w_e4, 256, 256, 4, 256, 0);
  wprep(ER1_C1_W, w_er1c1, 256, 256, 3, 256, 0);
  wprep(ER1_C2_W, w_er1c2, 256, 256, 1, 256, 0);
  wprep(ER2_C1_W, w_er2c1, 256, 256, 3, 256, 0);
  wprep(ER2_C2_W, w_er2c2, 256, 256, 1, 256, 0);
  wprep(DR1_C1_W, w_dr1c1, 256, 256, 3, 256, 0);
  wprep(DR1_C2_W, w_dr1c2, 256, 256, 1, 256, 0);
  wprep(DR2_C1_W, w_dr2c1, 256, 256, 3, 256, 0);
  wprep(DR2_C2_W, w_dr2c2, 256, 256, 1, 256, 0);
  wprep(D_CONVT1_W, w_d1, 256, 256, 4, 256, 1);
  wprep(D_CONVT2_W, w_d2, 128, 256, 3, 256, 1);
  wprep(D_CONVT3_W, w_d3, 64, 128, 3, 128, 1);
  wprep(D_CONVT4_W, w_d4, 3, 64, 4, 64, 1);
  k_prep_cb<<<dim3(512), 256, 0, stream>>>(P(CODEBOOK), (__bf16*)(ws + w_cb), wsq, 512, 256);

  auto conv = [&](const float* in, size_t w2off, int bi, float* out_,
                  int Nn, int Cin, int IH, int IW, int Cout, int OH, int OW,
                  int k, int s, int p, int d, int KcPad, int epi) {
    int M = Nn * OH * OW;
    int owLog = __builtin_ctz((unsigned)OW);
    int ohwLog = __builtin_ctz((unsigned)(OH * OW));
    dim3 grid((M + 127) / 128, (Cout + 63) / 64, 1);
    k_conv_wmma<<<grid, 256, 0, stream>>>(in, (const __bf16*)(ws + w2off), P(bi), out_,
                                          Nn, Cin, IH, IW, Cout, OH, OW,
                                          owLog, ohwLog, k, s, p, d, KcPad, epi);
  };
  auto bn = [&](const float* x, float* y, int gi, int bi,
                int Nn, int C, int HW, int prer, int postr) {
    k_bn_stats<<<dim3(C), 256, 0, stream>>>(x, stats, Nn, C, HW, prer);
    long tot = (long)Nn * C * HW;
    int blocks = (int)((tot + 255) / 256); if (blocks > 8192) blocks = 8192;
    k_bn_apply<<<dim3(blocks), 256, 0, stream>>>(x, y, stats, P(gi), P(bi),
                                                 tot, C, HW, prer, postr);
  };
  auto resblock = [&](const float* x, float* t1, float* t2, float* dest,
                      int bn1g, int bn1b, size_t w1, int c1b,
                      int bn2g, int bn2b, size_t w2, int c2b) {
    const int C = 256, HW = 1024, Nn = 32, S = 32;
    long tot = (long)Nn * C * HW;
    bn(x, t1, bn1g, bn1b, Nn, C, HW, 0, 1);
    conv(t1, w1, c1b, t2, Nn, C, S, S, C, S, S, 3, 1, 1, 1, 256, EPI_BIAS);
    bn(t2, t1, bn2g, bn2b, Nn, C, HW, 0, 1);
    conv(t1, w2, c2b, t2, Nn, C, S, S, C, S, S, 1, 1, 0, 1, 256, EPI_BIAS);
    k_add<<<dim3(4096), 256, 0, stream>>>(x, t2, dest, tot);
  };

  // -------- encoder --------
  conv(P(IN_X), w_e1, E_CONV1_B, bufA, 32, 3, 128, 128, 64, 64, 64, 4, 2, 1, 1, 32, EPI_RELU);
  bn(bufA, bufB, E_BN1_G, E_BN1_B, 32, 64, 4096, 0, 0);
  conv(bufB, w_e2, E_CONV2_B, bufA, 32, 64, 64, 64, 128, 64, 64, 3, 1, 1, 1, 64, EPI_RELU);
  bn(bufA, bufB, E_BN2_G, E_BN2_B, 32, 128, 4096, 0, 0);
  conv(bufB, w_e3, E_CONV3_B, bufA, 32, 128, 64, 64, 256, 64, 64, 3, 1, 1, 1, 128, EPI_RELU);
  bn(bufA, bufB, E_BN3_G, E_BN3_B, 32, 256, 4096, 0, 0);
  conv(bufB, w_e4, E_CONV4_B, bufA, 32, 256, 64, 64, 256, 32, 32, 4, 2, 1, 1, 256, EPI_BIAS);
  resblock(bufA, bufB, bufQ, bufB,
           ER1_BN1_G, ER1_BN1_B, w_er1c1, ER1_C1_B,
           ER1_BN2_G, ER1_BN2_B, w_er1c2, ER1_C2_B);
  resblock(bufB, bufA, bufQ, bufC,   // z -> bufC
           ER2_BN1_G, ER2_BN1_B, w_er2c1, ER2_C1_B,
           ER2_BN2_G, ER2_BN2_B, w_er2c2, ER2_C2_B);

  // -------- vector quantization --------
  k_vq_argmin<<<dim3(32768 / 16), 256, 0, stream>>>(
      bufC, (const __bf16*)(ws + w_cb), wsq, idxb, 32768, 256, 1024, 512);
  k_zero<<<dim3(1), 256, 0, stream>>>(loss, 1);
  k_vq_gather<<<dim3(4096), 256, 0, stream>>>(
      bufC, P(CODEBOOK), idxb, bufQ, loss, (long)32 * 256 * 1024, 256, 1024);

  // -------- decoder --------
  resblock(bufQ, bufA, bufB, bufA,
           DR1_BN1_G, DR1_BN1_B, w_dr1c1, DR1_C1_B,
           DR1_BN2_G, DR1_BN2_B, w_dr1c2, DR1_C2_B);
  resblock(bufA, bufB, bufC, bufB,
           DR2_BN1_G, DR2_BN1_B, w_dr2c1, DR2_C1_B,
           DR2_BN2_G, DR2_BN2_B, w_dr2c2, DR2_C2_B);
  bn(bufB, bufA, D_BN0_G, D_BN0_B, 32, 256, 1024, 1, 0);
  conv(bufA, w_d1, D_CONVT1_B, bufB, 32, 256, 32, 32, 256, 64, 64, 4, 1, 2, 2, 256, EPI_RELU);
  bn(bufB, bufA, D_BN1_G, D_BN1_B, 32, 256, 4096, 0, 0);
  conv(bufA, w_d2, D_CONVT2_B, bufB, 32, 256, 64, 64, 128, 64, 64, 3, 1, 1, 1, 256, EPI_RELU);
  bn(bufB, bufA, D_BN2_G, D_BN2_B, 32, 128, 4096, 0, 0);
  conv(bufA, w_d3, D_CONVT3_B, bufB, 32, 128, 64, 64, 64, 64, 64, 3, 1, 1, 1, 128, EPI_RELU);
  bn(bufB, bufA, D_BN3_G, D_BN3_B, 32, 64, 4096, 0, 0);
  conv(bufA, w_d4, D_CONVT4_B, outp, 32, 64, 64, 64, 3, 128, 128, 4, 1, 2, 2, 64, EPI_TANH);

  k_final<<<dim3(1), 1, 0, stream>>>(loss, outp + 32 * 3 * 128 * 128,
                                     2.0f / 8388608.0f);
}
